// ConvGraphQNN_65481071402317
// MI455X (gfx1250) — compile-verified, matrix operand order checked by reading, not run on hardware
//
#include <hip/hip_runtime.h>
#include <hip/hip_bf16.h>

// ---------------------------------------------------------------------------
// ConvGraphQNN on MI455X (gfx1250)
//
// Math reduction: features are 1-D and strictly positive (sigmoid), so the
// cosine-similarity graph is fully connected (sim ~= 1 >= 0.9 always):
//   out[b] = 2 * mean_{pixels}( sigmoid( patch . W + bias ) )
// The length-4 patch dot products are done with V_WMMA_F32_16X16X4_F32
// (16 pixels per WMMA per wave, exact f32 math).
// ---------------------------------------------------------------------------

typedef float    v2f  __attribute__((ext_vector_type(2)));
typedef float    v8f  __attribute__((ext_vector_type(8)));
typedef _Float16 v16h __attribute__((ext_vector_type(16)));

#define BATCH   2
#define IH      96
#define IW      96
#define HO      95
#define WO      95
#define LPB     (HO * WO)          // 9025 pixels per batch/graph
#define TOTAL   (BATCH * LPB)      // 18050
#define NTILES  ((TOTAL + 15) / 16)
#define NBLK    142                // blocks for the conv kernel
#define WPB     8                  // waves per block (256 threads, wave32)
#define NWAVES  (NBLK * WPB)

#if defined(__has_builtin)
#  if __has_builtin(__builtin_amdgcn_wmma_f32_16x16x4_f32)
#    define USE_WMMA_X4 1
#  endif
#endif

__device__ __forceinline__ float sigmoidf_fast(float z) {
    return 1.0f / (1.0f + __expf(-z));
}

__global__ __launch_bounds__(256) void conv_wmma_partial(
    const float* __restrict__ x,      // [2,1,96,96]
    const float* __restrict__ w,      // [1,4]
    const float* __restrict__ bptr,   // [1]
    float* __restrict__ partial)      // [NBLK][2]
{
    const int tid  = threadIdx.x;
    const int lane = tid & 31;
    const int m    = lane & 15;       // row within 16-pixel tile / column N
    const int half = lane >> 4;       // 0: K=0,1 (M rows 0-7) ; 1: K=2,3 (M rows 8-15)
    const int wave = tid >> 5;
    const int gwav = blockIdx.x * WPB + wave;

    const float w0 = w[0], w1 = w[1], w2 = w[2], w3 = w[3];
    const float bias = bptr[0];

    float acc0 = 0.0f;  // sum of activations, batch 0
    float acc1 = 0.0f;  // sum of activations, batch 1

    for (int t = gwav; t < NTILES; t += NWAVES) {
        const int base  = t * 16;
        const int p     = base + m;                 // this lane's A-row pixel
        const int valid = (p < TOTAL) ? 1 : 0;
        const int pc    = valid ? p : 0;            // clamp for safe addressing
        const int bsel  = (pc >= LPB) ? 1 : 0;
        const int pix   = pc - bsel * LPB;
        const int i     = pix / WO;
        const int j     = pix - i * WO;
        const int idx   = bsel * (IH * IW) + i * IW + j;

#ifdef USE_WMMA_X4
        // A layout (32-bit A 16x4): lane m holds {K0,K1}; lane m+16 holds {K2,K3}.
        // K -> patch element: K = 2*half + c  => (di = half, dj = c).
        float a0 = x[idx + half * IW];
        float a1 = x[idx + half * IW + 1];
        a0 = valid ? a0 : 0.0f;
        a1 = valid ? a1 : 0.0f;

        v2f A;  A[0] = a0;  A[1] = a1;
        // B (4x16): all 16 columns = W; VGPR0 = rows K0/K2, VGPR1 = rows K1/K3.
        v2f Bm; Bm[0] = half ? w2 : w0;  Bm[1] = half ? w3 : w1;
        v8f C = {};
        v8f D = __builtin_amdgcn_wmma_f32_16x16x4_f32(
                    false, A, false, Bm, (short)0, C, false, false);
#else
        // Fallback: f16 16x16x32, K zero-padded to 32. All four K values live
        // in half==0 lanes (A v16h elems 0..3 = K0..K3; B elems 0..3 = W).
        float a0 = x[idx];
        float a1 = x[idx + 1];
        float a2 = x[idx + IW];
        float a3 = x[idx + IW + 1];
        const int use = (valid && half == 0) ? 1 : 0;
        v16h A  = {};
        v16h Bm = {};
        A[0] = use ? (_Float16)a0 : (_Float16)0.0f;
        A[1] = use ? (_Float16)a1 : (_Float16)0.0f;
        A[2] = use ? (_Float16)a2 : (_Float16)0.0f;
        A[3] = use ? (_Float16)a3 : (_Float16)0.0f;
        Bm[0] = (half == 0) ? (_Float16)w0 : (_Float16)0.0f;
        Bm[1] = (half == 0) ? (_Float16)w1 : (_Float16)0.0f;
        Bm[2] = (half == 0) ? (_Float16)w2 : (_Float16)0.0f;
        Bm[3] = (half == 0) ? (_Float16)w3 : (_Float16)0.0f;
        v8f C = {};
        v8f D = __builtin_amdgcn_wmma_f32_16x16x32_f16(
                    false, A, false, Bm, (short)0, C, false, false);
#endif

        // D layout: VGPR r, lanes 0-15 -> M=r, lanes 16-31 -> M=8+r (col = m).
        // Every column is identical; take column 0 only (lanes 0 and 16).
        const int take = (m == 0) ? 1 : 0;
#pragma unroll
        for (int r = 0; r < 8; ++r) {
            const int   pm = base + (half << 3) + r;
            const int   ok = take && (pm < TOTAL);
            const float s  = sigmoidf_fast(D[r] + bias);
            const float sv = ok ? s : 0.0f;
            acc0 += (pm < LPB) ? sv : 0.0f;
            acc1 += (pm < LPB) ? 0.0f : sv;
        }
    }

    // wave32 reduction
#pragma unroll
    for (int off = 16; off > 0; off >>= 1) {
        acc0 += __shfl_xor(acc0, off, 32);
        acc1 += __shfl_xor(acc1, off, 32);
    }

    __shared__ float sm[WPB][2];
    if (lane == 0) { sm[wave][0] = acc0; sm[wave][1] = acc1; }
    __syncthreads();
    if (tid == 0) {
        float s0 = 0.0f, s1 = 0.0f;
#pragma unroll
        for (int k = 0; k < WPB; ++k) { s0 += sm[k][0]; s1 += sm[k][1]; }
        partial[blockIdx.x * 2 + 0] = s0;
        partial[blockIdx.x * 2 + 1] = s1;
    }
}

__global__ __launch_bounds__(64) void finalize_out(
    const float* __restrict__ partial,  // [NBLK][2]
    float* __restrict__ out)            // [2]
{
    __shared__ float sm[64];
    const int t = threadIdx.x;          // 64 threads; even lanes batch0, odd batch1
    const int c = t & 1;
    float s = 0.0f;
    for (int g = (t >> 1); g < NBLK; g += 32) s += partial[2 * g + c];
    sm[t] = s;
    // parity-preserving tree reduction (strides all even)
    for (int str = 32; str >= 2; str >>= 1) {
        __syncthreads();
        if (t < str) sm[t] += sm[t + str];
    }
    __syncthreads();
    if (t == 0) out[0] = 2.0f * sm[0] / (float)LPB;
    if (t == 1) out[1] = 2.0f * sm[1] / (float)LPB;
}

extern "C" void kernel_launch(void* const* d_in, const int* in_sizes, int n_in,
                              void* d_out, int out_size, void* d_ws, size_t ws_size,
                              hipStream_t stream) {
    const float* x = (const float*)d_in[0];   // [2,1,96,96] f32
    const float* W = (const float*)d_in[1];   // [1,4] f32
    const float* b = (const float*)d_in[2];   // [1] f32
    float* out     = (float*)d_out;           // [2] f32
    float* partial = (float*)d_ws;            // NBLK*2 floats (1136 B)

    conv_wmma_partial<<<NBLK, 256, 0, stream>>>(x, W, b, partial);
    finalize_out<<<1, 64, 0, stream>>>(partial, out);
}
